// ParallelDroplessMLP_2302102471514
// MI455X (gfx1250) — matrix operand order repaired and checked by reading, not compiled, not run on hardware
//
#include <hip/hip_runtime.h>
#include <hip/hip_bf16.h>
#include <math.h>

#define NE    8
#define TOPK  2
#define SLEN  2048
#define BSZ   2
#define HS    1024
#define FFN   4096
#define NTOK  (SLEN * BSZ)      // 4096 tokens
#define TROWS (NTOK * TOPK)     // 8192 routed rows
#define MAX_TILES 80            // >= ceil(8192/128) + 8 experts

typedef __attribute__((ext_vector_type(16))) __bf16 bf16x16;
typedef __attribute__((ext_vector_type(2)))  __bf16 bf16x2;
typedef __attribute__((ext_vector_type(8)))  float  f32x8;
typedef __attribute__((__vector_size__(16))) int    v4i;   // matches builtin param type

#define GLOBAL_AS __attribute__((address_space(1)))
#define LDS_AS    __attribute__((address_space(3)))

#if __has_builtin(__builtin_amdgcn_global_load_async_to_lds_b128)
#define HAVE_ASYNC_LDS 1
#else
#define HAVE_ASYNC_LDS 0
#endif

union Frag {
    bf16x16 v;
    struct { uint4 lo; uint4 hi; } q;   // elements 0..7 / 8..15 (16B each)
};

__device__ __forceinline__ unsigned short f2bf(float f) {
    __bf16 h = (__bf16)f;               // lowers to v_cvt (RNE)
    union { __bf16 h; unsigned short s; } u; u.h = h;
    return u.s;
}

__device__ __forceinline__ unsigned f2bf2(float a, float b) {
    bf16x2 v = { (__bf16)a, (__bf16)b };   // lowers to v_cvt_pk_bf16_f32
    union { bf16x2 v; unsigned u; } u; u.v = v;
    return u.u;
}

#if HAVE_ASYNC_LDS
__device__ __forceinline__ void async_ldsload_b128(const unsigned* g, unsigned* l) {
    __builtin_amdgcn_global_load_async_to_lds_b128(
        (GLOBAL_AS v4i*)g, (LDS_AS v4i*)l, 0, 0);
}
__device__ __forceinline__ void wait_async0() {
#if __has_builtin(__builtin_amdgcn_s_wait_asynccnt)
    __builtin_amdgcn_s_wait_asynccnt(0);
#else
    asm volatile("s_wait_asynccnt 0" ::: "memory");
#endif
}
#endif

__device__ __forceinline__ float gelu_exact(float v) {
    return 0.5f * v * (1.0f + erff(v * 0.70710678118654752f));
}

// --------------------------------------------------------------------------
// Kernel 0a: one-time fp32 -> bf16 conversion of x (8 MB result).
// --------------------------------------------------------------------------
__global__ __launch_bounds__(256) void conv_x(
    const float* __restrict__ x, unsigned* __restrict__ xb32)
{
    int i = blockIdx.x * 256 + threadIdx.x;        // pair index
    if (i < NTOK * HS / 2) {
        float2 v = *(const float2*)&x[2 * i];
        xb32[i] = f2bf2(v.x, v.y);
    }
}

// --------------------------------------------------------------------------
// Kernel 0b: tiled convert+transpose: fp32 [E][K][N] -> packed bf16 pairs
// [E][N][K/2] (u32 = {k even lo, k odd hi}) — the exact B-fragment layout.
// Grid: (N/64, K/64, E), 256 threads.
// --------------------------------------------------------------------------
__global__ __launch_bounds__(256) void conv_transpose_w(
    const float* __restrict__ src, unsigned* __restrict__ dst, int K, int N)
{
    __shared__ float tile[64][65];
    const int e = blockIdx.z;
    const int kBase = blockIdx.y * 64;
    const int nBase = blockIdx.x * 64;
    const float* s = src + ((size_t)e * K + kBase) * N + nBase;
    unsigned*    d = dst + ((size_t)e * N + nBase) * (K / 2) + kBase / 2;
    const int tid = threadIdx.x;
#pragma unroll
    for (int i = 0; i < 16; ++i) {
        int p = i * 256 + tid;               // 4096 elements
        int k = p >> 6, n = p & 63;
        tile[k][n] = s[(size_t)k * N + n];   // coalesced along n
    }
    __syncthreads();
#pragma unroll
    for (int i = 0; i < 8; ++i) {
        int p = i * 256 + tid;               // 2048 output pairs
        int n = p >> 5, kp = p & 31;
        d[(size_t)n * (K / 2) + kp] = f2bf2(tile[2 * kp][n], tile[2 * kp + 1][n]);
    }
}

// --------------------------------------------------------------------------
// Kernel 1: stable counting sort of expert assignments + row-tile table.
// meta (ints): [0]=num_tiles; tiles at [16+4*i] = {rowStart, rowEnd, expert, pad}
// --------------------------------------------------------------------------
__global__ __launch_bounds__(256) void moe_sort(
    const int*   __restrict__ eidx,   // [TROWS]
    const float* __restrict__ ew,     // [TROWS]
    int*   __restrict__ meta,
    int*   __restrict__ row_token,    // [TROWS]
    float* __restrict__ row_scale)    // [TROWS]
{
    __shared__ int hist[256][NE];
    __shared__ int totals[NE];
    __shared__ int bins[NE + 1];
    const int t = threadIdx.x;
    const int per = TROWS / 256;      // 32
    const int base = t * per;

    int lc[NE];
#pragma unroll
    for (int e = 0; e < NE; ++e) lc[e] = 0;
    for (int j = 0; j < per; ++j) lc[eidx[base + j] & (NE - 1)]++;
#pragma unroll
    for (int e = 0; e < NE; ++e) hist[t][e] = lc[e];
    __syncthreads();

    if (t < NE) {               // exclusive scan of column t
        int run = 0;
        for (int i = 0; i < 256; ++i) { int c = hist[i][t]; hist[i][t] = run; run += c; }
        totals[t] = run;
    }
    __syncthreads();

    if (t == 0) {
        int acc = 0;
#pragma unroll
        for (int e = 0; e < NE; ++e) { bins[e] = acc; acc += totals[e]; }
        bins[NE] = acc;
        int nt = 0;
        for (int e = 0; e < NE; ++e)
            for (int rs = bins[e]; rs < bins[e + 1]; rs += 128) {
                meta[16 + nt * 4 + 0] = rs;
                meta[16 + nt * 4 + 1] = bins[e + 1];
                meta[16 + nt * 4 + 2] = e;
                nt++;
            }
        meta[0] = nt;
    }
    __syncthreads();

    int lr[NE];
#pragma unroll
    for (int e = 0; e < NE; ++e) lr[e] = 0;
    for (int j = 0; j < per; ++j) {
        int i = base + j;
        int e = eidx[i] & (NE - 1);
        int pos = bins[e] + hist[t][e] + lr[e];
        lr[e]++;
        row_token[pos] = i / TOPK;
        row_scale[pos] = ew[i];
    }
}

// Fragment layouts (ISA 7.12.2), staged tiles padded to 20 dwords/row:
//  As32[buf][r][kp]: row r, packed bf16 pair (k=2kp,2kp+1). Lane m row; lo=dw[half*4..+4),
//    hi=dw[8+half*4..+4). Bs32[buf][n][kp]: transposed; lane n=lane&15, lo=dw[half*4..+4), hi=dw[8+half*4..+4).
#define WMMA_BF16(ACC, A, B) \
    ACC = __builtin_amdgcn_wmma_f32_16x16x32_bf16(false, (A).v, false, (B).v, (short)0, ACC, false, false)

// Shared GEMM inner machinery: block tile 128x128, 8 waves of 32x64, K-step 32.
// Async path: both A and B tiles via global_load_async_to_lds_b128 (ping-pong),
// ONE barrier per K-step. Fallback: register staging, two barriers.
#define GEMM_BODY(KDIM, A_ROWPTR, B_ROWPTR)                                         \
    int buf = 0;                                                                    \
    for (int k0 = 0; k0 < (KDIM); k0 += 32) {                                       \
        PIPE_TOP(buf);                                                              \
        __syncthreads();                                                            \
        if (k0 + 32 < (KDIM)) { PIPE_NEXT(buf, k0 + 32); }                          \
        Frag a0, a1, b[4];                                                          \
        a0.q.lo = *(const uint4*)&As32[buf][rb + m][half * 4];                      \
        a0.q.hi = *(const uint4*)&As32[buf][rb + m][8 + half * 4];                  \
        a1.q.lo = *(const uint4*)&As32[buf][rb + 16 + m][half * 4];                 \
        a1.q.hi = *(const uint4*)&As32[buf][rb + 16 + m][8 + half * 4];             \
        _Pragma("unroll")                                                           \
        for (int ci = 0; ci < 4; ++ci) {                                            \
            b[ci].q.lo = *(const uint4*)&Bs32[buf][cb + ci * 16 + m][half * 4];     \
            b[ci].q.hi = *(const uint4*)&Bs32[buf][cb + ci * 16 + m][8 + half * 4]; \
        }                                                                           \
        _Pragma("unroll")                                                           \
        for (int ci = 0; ci < 4; ++ci) {                                            \
            WMMA_BF16(acc0[ci], a0, b[ci]);                                         \
            WMMA_BF16(acc1[ci], a1, b[ci]);                                         \
        }                                                                           \
        PIPE_BOT();                                                                 \
        buf ^= 1;                                                                   \
    }

// --------------------------------------------------------------------------
// Kernel 2: H = gelu(gather(xb) @ w1t[e]) -> bf16 workspace.
// --------------------------------------------------------------------------
__global__ __launch_bounds__(256) void moe_gemm1(
    const unsigned* __restrict__ xb32,    // [NTOK, HS/2] packed bf16 pairs
    const unsigned* __restrict__ w1t,     // [NE, FFN, HS/2] transposed bf16 pairs
    const int*   __restrict__ meta,
    const int*   __restrict__ row_token,
    unsigned short* __restrict__ H)       // [TROWS, FFN] bf16
{
    const int tileIdx = blockIdx.y;
    if (tileIdx >= meta[0]) return;
    const int rowStart = meta[16 + tileIdx * 4 + 0];
    const int rowEnd   = meta[16 + tileIdx * 4 + 1];
    const int expert   = meta[16 + tileIdx * 4 + 2];
    const int colBase  = blockIdx.x * 128;

    __shared__ unsigned As32[2][128][20];
    __shared__ unsigned Bs32[2][128][20];
    __shared__ int tok[128];

    const int tid = threadIdx.x;
    if (tid < 128) {
        int r = rowStart + tid;
        tok[tid] = (r < rowEnd) ? row_token[r] : 0;
    }
    __syncthreads();

    const int lane = tid & 31;
    const int wave = tid >> 5;
    const int rb = (wave & 3) * 32;
    const int cb = (wave >> 2) * 64;
    const int m = lane & 15;
    const int half = lane >> 4;

    f32x8 acc0[4] = {}, acc1[4] = {};
    const unsigned* w1te = w1t + (size_t)expert * FFN * (HS / 2);

#if HAVE_ASYNC_LDS
    auto issueTiles = [&](int bu, int kk) {
#pragma unroll
        for (int i = 0; i < 2; ++i) {
            int p = i * 256 + tid;
            int r = p >> 2, q = p & 3;                      // 4 x 16B chunks per row
            async_ldsload_b128(xb32 + (size_t)tok[r] * (HS / 2) + (kk >> 1) + 4 * q,
                               &As32[bu][r][4 * q]);
        }
#pragma unroll
        for (int i = 0; i < 2; ++i) {
            int p = i * 256 + tid;
            int n = p >> 2, q = p & 3;
            async_ldsload_b128(w1te + (size_t)(colBase + n) * (HS / 2) + (kk >> 1) + 4 * q,
                               &Bs32[bu][n][4 * q]);
        }
    };
    issueTiles(0, 0);
#define PIPE_TOP(BUF)        wait_async0()
#define PIPE_NEXT(BUF, KK)   issueTiles((BUF) ^ 1, (KK))
#define PIPE_BOT()
#else
    unsigned aP[8], bP[8];
    auto loadTiles = [&](int kk) {
#pragma unroll
        for (int i = 0; i < 8; ++i) {
            int p = i * 256 + tid;
            aP[i] = xb32[(size_t)tok[p >> 4] * (HS / 2) + (kk >> 1) + (p & 15)];
            bP[i] = w1te[(size_t)(colBase + (p >> 4)) * (HS / 2) + (kk >> 1) + (p & 15)];
        }
    };
    auto stageTiles = [&](int bu) {
#pragma unroll
        for (int i = 0; i < 8; ++i) {
            int p = i * 256 + tid;
            As32[bu][p >> 4][p & 15] = aP[i];
            Bs32[bu][p >> 4][p & 15] = bP[i];
        }
    };
    loadTiles(0);
#define PIPE_TOP(BUF)        stageTiles(BUF)
#define PIPE_NEXT(BUF, KK)   loadTiles(KK)
#define PIPE_BOT()           __syncthreads()
#endif

    GEMM_BODY(HS, /*A*/, /*B*/)

#undef PIPE_TOP
#undef PIPE_NEXT
#undef PIPE_BOT

    const int nn = lane & 15;
    const int mofs = (lane >> 4) * 8;
#pragma unroll
    for (int ci = 0; ci < 4; ++ci) {
#pragma unroll
        for (int i = 0; i < 8; ++i) {
            int mm = mofs + i;
            int r0 = rowStart + rb + mm;
            int r1 = r0 + 16;
            int col = colBase + cb + ci * 16 + nn;
            if (r0 < rowEnd) H[(size_t)r0 * FFN + col] = f2bf(gelu_exact(acc0[ci][i]));
            if (r1 < rowEnd) H[(size_t)r1 * FFN + col] = f2bf(gelu_exact(acc1[ci][i]));
        }
    }
}

// --------------------------------------------------------------------------
// Kernel 3: y = H @ w2t[e]; atomic scatter-combine into out and expert_output.
// --------------------------------------------------------------------------
__global__ __launch_bounds__(256) void moe_gemm2(
    const unsigned* __restrict__ H32,       // [TROWS, FFN/2] packed bf16 pairs
    const unsigned* __restrict__ w2t,       // [NE, HS, FFN/2] transposed bf16 pairs
    const int*   __restrict__ meta,
    const int*   __restrict__ row_token,
    const float* __restrict__ row_scale,
    float* __restrict__ out,                // [NTOK, HS]
    float* __restrict__ eout)               // [BSZ, NE, SLEN, HS]
{
    const int tileIdx = blockIdx.y;
    if (tileIdx >= meta[0]) return;
    const int rowStart = meta[16 + tileIdx * 4 + 0];
    const int rowEnd   = meta[16 + tileIdx * 4 + 1];
    const int expert   = meta[16 + tileIdx * 4 + 2];
    const int colBase  = blockIdx.x * 128;

    __shared__ unsigned As32[2][128][20];
    __shared__ unsigned Bs32[2][128][20];
    __shared__ int   tok[128];
    __shared__ float scl[128];

    const int tid = threadIdx.x;
    if (tid < 128) {
        int r = rowStart + tid;
        bool v = r < rowEnd;
        tok[tid] = v ? row_token[r] : 0;
        scl[tid] = v ? row_scale[r] : 0.f;
    }
    __syncthreads();

    const int lane = tid & 31;
    const int wave = tid >> 5;
    const int rb = (wave & 3) * 32;
    const int cb = (wave >> 2) * 64;
    const int m = lane & 15;
    const int half = lane >> 4;

    f32x8 acc0[4] = {}, acc1[4] = {};
    const unsigned* w2te = w2t + (size_t)expert * HS * (FFN / 2);

#if HAVE_ASYNC_LDS
    auto issueTiles = [&](int bu, int kk) {
#pragma unroll
        for (int i = 0; i < 2; ++i) {
            int p = i * 256 + tid;
            int r = p >> 2, q = p & 3;
            int gr = rowStart + r; if (gr > TROWS - 1) gr = TROWS - 1;
            async_ldsload_b128(H32 + (size_t)gr * (FFN / 2) + (kk >> 1) + 4 * q,
                               &As32[bu][r][4 * q]);
        }
#pragma unroll
        for (int i = 0; i < 2; ++i) {
            int p = i * 256 + tid;
            int n = p >> 2, q = p & 3;
            async_ldsload_b128(w2te + (size_t)(colBase + n) * (FFN / 2) + (kk >> 1) + 4 * q,
                               &Bs32[bu][n][4 * q]);
        }
    };
    issueTiles(0, 0);
#define PIPE_TOP(BUF)        wait_async0()
#define PIPE_NEXT(BUF, KK)   issueTiles((BUF) ^ 1, (KK))
#define PIPE_BOT()
#else
    unsigned aP[8], bP[8];
    auto loadTiles = [&](int kk) {
#pragma unroll
        for (int i = 0; i < 8; ++i) {
            int p = i * 256 + tid;
            int gr = rowStart + (p >> 4); if (gr > TROWS - 1) gr = TROWS - 1;
            aP[i] = H32[(size_t)gr * (FFN / 2) + (kk >> 1) + (p & 15)];
            bP[i] = w2te[(size_t)(colBase + (p >> 4)) * (FFN / 2) + (kk >> 1) + (p & 15)];
        }
    };
    auto stageTiles = [&](int bu) {
#pragma unroll
        for (int i = 0; i < 8; ++i) {
            int p = i * 256 + tid;
            As32[bu][p >> 4][p & 15] = aP[i];
            Bs32[bu][p >> 4][p & 15] = bP[i];
        }
    };
    loadTiles(0);
#define PIPE_TOP(BUF)        stageTiles(BUF)
#define PIPE_NEXT(BUF, KK)   loadTiles(KK)
#define PIPE_BOT()           __syncthreads()
#endif

    GEMM_BODY(FFN, /*A*/, /*B*/)

#undef PIPE_TOP
#undef PIPE_NEXT
#undef PIPE_BOT

    const int nn = lane & 15;
    const int mofs = (lane >> 4) * 8;

    auto emit = [&](float v, int rl, int col) {
        int r = rowStart + rl;
        if (r < rowEnd) {
            int t = tok[rl];
            float s = scl[rl];
            atomicAdd(&out[(size_t)t * HS + col], s * v);
            int si = t / BSZ, b = t % BSZ;   // token = s*BS + b
            atomicAdd(&eout[(((size_t)b * NE + expert) * SLEN + si) * HS + col], v);
        }
    };

#pragma unroll
    for (int ci = 0; ci < 4; ++ci) {
#pragma unroll
        for (int i = 0; i < 8; ++i) {
            int mm = mofs + i;
            int col = colBase + cb + ci * 16 + nn;
            emit(acc0[ci][i], rb + mm,      col);
            emit(acc1[ci][i], rb + 16 + mm, col);
        }
    }
}

// --------------------------------------------------------------------------
extern "C" void kernel_launch(void* const* d_in, const int* in_sizes, int n_in,
                              void* d_out, int out_size, void* d_ws, size_t ws_size,
                              hipStream_t stream) {
    (void)in_sizes; (void)n_in; (void)ws_size;

    const float* x    = (const float*)d_in[0];   // [SL, BS, HS]
    const float* ew   = (const float*)d_in[1];   // [NTOK, TOPK]
    const float* w1   = (const float*)d_in[2];   // [NE, HS, FFN]
    const float* w2   = (const float*)d_in[3];   // [NE, FFN, HS]
    const int*   eidx = (const int*)d_in[4];     // [NTOK, TOPK]

    float* out  = (float*)d_out;                       // [NTOK, HS]
    float* eout = out + (size_t)NTOK * HS;             // [BSZ, NE, SLEN, HS]

    char* ws = (char*)d_ws;
    int*      meta      = (int*)ws;                        // 8 KB
    int*      row_token = (int*)(ws + 8 * 1024);           // 32 KB
    float*    row_scale = (float*)(ws + 40 * 1024);        // 32 KB
    unsigned* xb32      = (unsigned*)(ws + 1024 * 1024);   // 8 MB bf16 x
    unsigned* wt        = (unsigned*)(ws + 16ull * 1024 * 1024);  // 64 MB: w1t, then w2t
    unsigned short* H   = (unsigned short*)(ws + 80ull * 1024 * 1024);  // 64 MB bf16

    (void)hipMemsetAsync(d_out, 0, (size_t)out_size * sizeof(float), stream);
    conv_x<<<(NTOK * HS / 2 + 255) / 256, 256, 0, stream>>>(x, xb32);
    moe_sort<<<1, 256, 0, stream>>>(eidx, ew, meta, row_token, row_scale);
    // w1 [E][HS][FFN] -> w1t [E][FFN][HS/2]
    conv_transpose_w<<<dim3(FFN / 64, HS / 64, NE), 256, 0, stream>>>(w1, wt, HS, FFN);
    moe_gemm1<<<dim3(FFN / 128, MAX_TILES), 256, 0, stream>>>(xb32, wt, meta, row_token, H);
    // w2 [E][FFN][HS] -> w2t [E][HS][FFN/2] (reuses the same buffer, stream-ordered)
    conv_transpose_w<<<dim3(HS / 64, FFN / 64, NE), 256, 0, stream>>>(w2, wt, FFN, HS);
    moe_gemm2<<<dim3(HS / 128, MAX_TILES), 256, 0, stream>>>((const unsigned*)H, wt, meta, row_token,
                                                             row_scale, out, eout);
}
